// RotaryCrossAttention_51410758533607
// MI455X (gfx1250) — matrix operand order repaired and checked by reading, not compiled, not run on hardware
//
#include <hip/hip_runtime.h>
#include <hip/hip_bf16.h>

// ---------------------------------------------------------------------------
// RotaryCrossAttention on MI455X (gfx1250): bf16 WMMA pipeline + TDM staging.
// B=2, NQ=2048, NK=8192, D=512, H=8, DH=64
// ---------------------------------------------------------------------------

#define BB  2
#define NQ  2048
#define NK  8192
#define DD  512
#define HH  8
#define DH  64

typedef __attribute__((ext_vector_type(16))) __bf16 v16bf;
typedef __attribute__((ext_vector_type(8)))  float  v8f;
typedef __attribute__((ext_vector_type(4)))  unsigned int u32x4;
typedef __attribute__((ext_vector_type(8)))  int i32x8;
typedef __attribute__((ext_vector_type(4)))  int i32x4;

union BFrag { v16bf v; unsigned short u[16]; };
union FAcc  { v8f  v; float f[8]; };

__device__ __forceinline__ unsigned short f2bf(float f) {
  unsigned int u = __builtin_bit_cast(unsigned int, f);
  unsigned int r = u + 0x7FFFu + ((u >> 16) & 1u);   // round-to-nearest-even
  return (unsigned short)(r >> 16);
}

// ---------------------------------------------------------------------------
// Tensor Data Mover: issue one D# (groups 2/3 zero -> <=2D tensor).
// Group0: count=1 | lds_addr | global_addr[56:0] | type=2.
// Group1 words w0..w5 carry data_size/dims/strides (see CDNA5 ISA).
// ---------------------------------------------------------------------------
__device__ __forceinline__ void tdm_load_2d(unsigned int lds_off,
                                            unsigned long long gaddr,
                                            int w0, int w1, int w2, int w3,
                                            int w4, int w5) {
  u32x4 g0 = { 1u,                                   // count=1, user mode
               lds_off,                              // LDS byte address
               (unsigned int)gaddr,                  // global_addr[31:0]
               (unsigned int)(gaddr >> 32) | (2u << 30) };  // [56:32] | type=2
  i32x8 g1 = { w0, w1, w2, w3, w4, w5, 0, 0 };
  i32x4 z4 = { 0, 0, 0, 0 };
#if defined(__clang_major__) && __clang_major__ >= 23
  i32x8 z8 = { 0, 0, 0, 0, 0, 0, 0, 0 };
  __builtin_amdgcn_tensor_load_to_lds(g0, g1, z4, z4, z8, 0);
#else
  __builtin_amdgcn_tensor_load_to_lds(g0, g1, z4, z4, 0);
#endif
}

// ---------------------------------------------------------------------------
// LayerNorm: one wave32 per 512-element row, f32 in -> bf16 out
// ---------------------------------------------------------------------------
__global__ void ln_bf16_kernel(const float* __restrict__ x,
                               const float* __restrict__ w,
                               const float* __restrict__ bsh,
                               unsigned short* __restrict__ out) {
  const int wave = threadIdx.x >> 5, lane = threadIdx.x & 31;
  const int row  = blockIdx.x * 8 + wave;
  const float* xr = x + (size_t)row * DD;
  float v[16];
  float s = 0.f;
#pragma unroll
  for (int i = 0; i < 4; ++i) {
    float4 t = ((const float4*)xr)[i * 32 + lane];
    v[i*4+0] = t.x; v[i*4+1] = t.y; v[i*4+2] = t.z; v[i*4+3] = t.w;
    s += t.x + t.y + t.z + t.w;
  }
#pragma unroll
  for (int off = 1; off < 32; off <<= 1) s += __shfl_xor(s, off, 32);
  const float mu = s * (1.0f / DD);
  float q = 0.f;
#pragma unroll
  for (int j = 0; j < 16; ++j) { float d = v[j] - mu; q += d * d; }
#pragma unroll
  for (int off = 1; off < 32; off <<= 1) q += __shfl_xor(q, off, 32);
  const float rs = rsqrtf(q * (1.0f / DD) + 1e-5f);
#pragma unroll
  for (int i = 0; i < 4; ++i) {
    int d0 = (i * 32 + lane) * 4;
    unsigned int p0 = (unsigned int)f2bf((v[i*4+0]-mu)*rs*w[d0+0] + bsh[d0+0]) |
                      ((unsigned int)f2bf((v[i*4+1]-mu)*rs*w[d0+1] + bsh[d0+1]) << 16);
    unsigned int p1 = (unsigned int)f2bf((v[i*4+2]-mu)*rs*w[d0+2] + bsh[d0+2]) |
                      ((unsigned int)f2bf((v[i*4+3]-mu)*rs*w[d0+3] + bsh[d0+3]) << 16);
    ((unsigned int*)(out + (size_t)row * DD))[d0/2]     = p0;
    ((unsigned int*)(out + (size_t)row * DD))[d0/2 + 1] = p1;
  }
}

// ---------------------------------------------------------------------------
__global__ void cvt_bf16_kernel(const float* __restrict__ in,
                                unsigned short* __restrict__ out, int n) {
  int i = blockIdx.x * 256 + threadIdx.x;
  if (i < n) out[i] = f2bf(in[i]);
}

// ---------------------------------------------------------------------------
// bf16 WMMA GEMM, 128x64 tile per 256-thread block, K staged via LDS.
// W tile staged TRANSPOSED ([n][k]) so B-fragment gathers are contiguous.
// MODE 0: Q proj  -> RoPE + 1/sqrt(DH), head-major bf16 [b,h,tok,d]
// MODE 1: KV proj -> n<512: RoPE->K [b,h,tok,d]; n>=512: V TRANSPOSED [b,h,d,tok]
// MODE 2: out proj -> f32 + bias
// ---------------------------------------------------------------------------
#define TM 128
#define TN 64
#define TK 32

template<int MODE>
__global__ void gemm_bf16_kernel(const unsigned short* __restrict__ A,   // M x 512
                                 const unsigned short* __restrict__ W,   // 512 x N
                                 int N, int TOK,
                                 const float* __restrict__ rope,
                                 unsigned short* __restrict__ outA,
                                 unsigned short* __restrict__ outV,
                                 const float* __restrict__ bias,
                                 float* __restrict__ outF) {
  __shared__ unsigned short Asm[TM * TK];   // 8 KB  [row][k]
  __shared__ unsigned short Wsm[TN * TK];   // 4 KB  [n][k]  (transposed)
  __shared__ float          Osm[TM * TN];   // 32 KB
  const int tid = threadIdx.x, wave = tid >> 5, lane = tid & 31;
  const int lrow = lane & 15, lhi = lane >> 4;
  const int m0 = blockIdx.x * TM, n0 = blockIdx.y * TN;

  FAcc acc[4];
#pragma unroll
  for (int t = 0; t < 4; ++t)
#pragma unroll
    for (int e = 0; e < 8; ++e) acc[t].f[e] = 0.f;

  for (int k0 = 0; k0 < DD; k0 += TK) {
    __syncthreads();
    // stage A tile (128x32) -- coalesced dword copies
    for (int i = tid; i < (TM*TK)/2; i += 256) {
      int r = i >> 4, c2 = i & 15;
      ((unsigned int*)Asm)[i] =
        *(const unsigned int*)(A + (size_t)(m0 + r) * DD + k0 + c2 * 2);
    }
    // stage W tile (32x64) transposed: coalesced read, [n][k] write
    for (int i = tid; i < (TK*TN)/2; i += 256) {
      int kk = i >> 5, n2 = i & 31;
      unsigned int p = *(const unsigned int*)(W + (size_t)(k0 + kk) * N + n0 + n2 * 2);
      Wsm[(n2*2    ) * TK + kk] = (unsigned short)(p & 0xFFFFu);
      Wsm[(n2*2 + 1) * TK + kk] = (unsigned short)(p >> 16);
    }
    if (k0 + TK < DD)   // global_prefetch_b8 of next A tile
      __builtin_prefetch(A + (size_t)(m0 + (tid >> 1)) * DD + k0 + TK + (tid & 1) * 16, 0, 0);
    __syncthreads();
    // A fragment (16x32 bf16) per ISA layout
    BFrag a;
    const int arow = wave * 16 + lrow, kb = lhi * 8;
#pragma unroll
    for (int e = 0; e < 8; ++e) {
      a.u[e]     = Asm[arow * TK + kb + e];
      a.u[e + 8] = Asm[arow * TK + kb + 16 + e];
    }
#pragma unroll
    for (int t = 0; t < 4; ++t) {
      BFrag bf;
      const int ncol = t * 16 + lrow, kbb = lhi * 8;
#pragma unroll
      for (int e = 0; e < 8; ++e) {
        bf.u[e]     = Wsm[ncol * TK + kbb + e];
        bf.u[e + 8] = Wsm[ncol * TK + kbb + 16 + e];
      }
      acc[t].v = __builtin_amdgcn_wmma_f32_16x16x32_bf16(
          false, a.v, false, bf.v, (short)0, acc[t].v, false, false);
    }
  }

  // dump this wave's 16x64 result to its private LDS strip
  float* Ow = Osm + wave * (16 * 64);
#pragma unroll
  for (int t = 0; t < 4; ++t)
#pragma unroll
    for (int r = 0; r < 8; ++r)
      Ow[(r + 8 * lhi) * 64 + t * 16 + lrow] = acc[t].f[r];
  asm volatile("s_wait_dscnt 0x0" ::: "memory");

  if (MODE == 1 && n0 >= 512) {
    // V: write transposed [b,h,d,tok]; lanes walk tokens for coalescing
    const int h = (n0 - 512) >> 6;
    for (int idx = lane; idx < 16 * 64; idx += 32) {
      const int r = idx & 15, d = idx >> 4;
      const int m = m0 + wave * 16 + r;
      const int b = m / TOK, tok = m % TOK;
      outV[((size_t)(b * HH + h) * DH + d) * TOK + tok] = f2bf(Ow[r * 64 + d]);
    }
  } else {
    for (int idx = lane; idx < 16 * 64; idx += 32) {
      const int r = idx >> 6, d = idx & 63;
      const int m = m0 + wave * 16 + r;
      float x = Ow[idx];
      if (MODE == 2) {
        outF[(size_t)m * DD + n0 + d] = x + bias[n0 + d];
      } else {
        const int b = m / TOK, tok = m % TOK;
        const float f  = rope[(size_t)m * DH + d];
        const float rh = (d < 32) ? -Ow[(r << 6) + 2 * d + 1]
                                  :  Ow[(r << 6) + 2 * (d - 32)];
        x = x * __cosf(f) + rh * __sinf(f);
        if (MODE == 0) x *= 0.125f;            // fold in 1/sqrt(DH)
        const int h = n0 >> 6;
        outA[((size_t)(b * HH + h) * TOK + tok) * DH + d] = f2bf(x);
      }
    }
  }
}

// ---------------------------------------------------------------------------
// Flash attention: per (b,h), 128 queries/block (16 per wave), 64-key tiles.
// K [key][dim] and V^T [dim][key] tiles double-buffered in LDS via the
// Tensor Data Mover; bf16 K/V (32MB total) stays resident in the 192MB L2.
// ---------------------------------------------------------------------------
__global__ void flash_kernel(const unsigned short* __restrict__ Q,   // [b,h,q,d]
                             const unsigned short* __restrict__ K,   // [b,h,k,d]
                             const unsigned short* __restrict__ V,   // [b,h,d,k]
                             unsigned short* __restrict__ Oattn) {
  __shared__ unsigned short Ksm[2][64 * 64];     // 2 x 8 KB
  __shared__ unsigned short Vsm[2][64 * 64];     // 2 x 8 KB (transposed tile)
  __shared__ unsigned short Psm[8 * 16 * 64];    // 16 KB per-wave P scratch
  const int tid = threadIdx.x, wave = tid >> 5, lane = tid & 31;
  const int lrow = lane & 15, lhi = lane >> 4;
  const int bh = blockIdx.x, b = bh >> 3, h = bh & 7;
  const int q0 = blockIdx.y * 128 + wave * 16;

  // Q fragments (16 queries x 64 dims = two K=32 A-fragments), uint4 loads
  const unsigned short* Qbase = Q + ((size_t)bh * NQ + q0) * DH;
  BFrag qf[2];
#pragma unroll
  for (int c = 0; c < 2; ++c) {
    const unsigned short* qp = Qbase + lrow * DH + c * 32 + lhi * 8;
    *(uint4*)&qf[c].u[0] = *(const uint4*)qp;
    *(uint4*)&qf[c].u[8] = *(const uint4*)(qp + 16);
  }

  float mrow[8], lsum[8];
  FAcc oacc[4];
#pragma unroll
  for (int r = 0; r < 8; ++r) { mrow[r] = -1e30f; lsum[r] = 0.f; }
#pragma unroll
  for (int t = 0; t < 4; ++t)
#pragma unroll
    for (int e = 0; e < 8; ++e) oacc[t].f[e] = 0.f;

  const unsigned short* Kb = K + (size_t)bh * NK * DH;
  const unsigned short* Vb = V + (size_t)bh * DH * NK;
  const int NT = NK / 64;

  // TDM tile issue: K tile = contiguous 2048 dwords; V^T tile = 2D 32x64 dwords
  auto issue_tile = [&](int kt, int buf) {
    tdm_load_2d((unsigned int)(size_t)&Ksm[buf][0],
                (unsigned long long)(size_t)Kb + (size_t)kt * 8192,
                0x00020000, 0x08000000, 0x00010000, 0x08000000, 1, 2048);
    tdm_load_2d((unsigned int)(size_t)&Vsm[buf][0],
                (unsigned long long)(size_t)Vb + (size_t)kt * 128,
                0x00020000, 0x10000000, 0x00400000, 0x00200000, 64, 4096);
  };
  if (wave == 0) issue_tile(0, 0);

  for (int kt = 0; kt < NT; ++kt) {
    __syncthreads();   // all waves done reading buffer (kt+1)&1 from tile kt-1
    if (wave == 0) {
      if (kt + 1 < NT) {
        issue_tile(kt + 1, (kt + 1) & 1);
        __builtin_amdgcn_s_wait_tensorcnt(2);   // tile kt's 2 loads complete
      } else {
        __builtin_amdgcn_s_wait_tensorcnt(0);
      }
    }
    __syncthreads();   // buffer kt&1 is ready for everyone
    const unsigned short* Kt = Ksm[kt & 1];
    const unsigned short* Vt = Vsm[kt & 1];

    // S = Q @ K^T  (16 queries x 64 keys)
    FAcc sacc[4];
#pragma unroll
    for (int t = 0; t < 4; ++t)
#pragma unroll
      for (int e = 0; e < 8; ++e) sacc[t].f[e] = 0.f;
#pragma unroll
    for (int t = 0; t < 4; ++t)
#pragma unroll
      for (int c = 0; c < 2; ++c) {
        BFrag kf;
        const int key = t * 16 + lrow, db = c * 32 + lhi * 8;
#pragma unroll
        for (int e = 0; e < 8; ++e) {
          kf.u[e]     = Kt[key * DH + db + e];
          kf.u[e + 8] = Kt[key * DH + db + 16 + e];
        }
        sacc[t].v = __builtin_amdgcn_wmma_f32_16x16x32_bf16(
            false, qf[c].v, false, kf.v, (short)0, sacc[t].v, false, false);
      }

    // online softmax (row = acc element r + 8*lhi, lives on a 16-lane group)
    float mt[8];
#pragma unroll
    for (int r = 0; r < 8; ++r) {
      mt[r] = fmaxf(fmaxf(sacc[0].f[r], sacc[1].f[r]),
                    fmaxf(sacc[2].f[r], sacc[3].f[r]));
    }
#pragma unroll
    for (int off = 1; off < 16; off <<= 1)
#pragma unroll
      for (int r = 0; r < 8; ++r) mt[r] = fmaxf(mt[r], __shfl_xor(mt[r], off, 32));
    float scl[8], ps[8];
#pragma unroll
    for (int r = 0; r < 8; ++r) {
      const float mn = fmaxf(mrow[r], mt[r]);
      scl[r] = __expf(mrow[r] - mn);
      mrow[r] = mn;
      ps[r] = 0.f;
    }
#pragma unroll
    for (int t = 0; t < 4; ++t)
#pragma unroll
      for (int r = 0; r < 8; ++r) {
        const float p = __expf(sacc[t].f[r] - mrow[r]);
        sacc[t].f[r] = p;
        ps[r] += p;
      }
#pragma unroll
    for (int off = 1; off < 16; off <<= 1)
#pragma unroll
      for (int r = 0; r < 8; ++r) ps[r] += __shfl_xor(ps[r], off, 32);
#pragma unroll
    for (int r = 0; r < 8; ++r) lsum[r] = lsum[r] * scl[r] + ps[r];
#pragma unroll
    for (int t = 0; t < 4; ++t)
#pragma unroll
      for (int r = 0; r < 8; ++r) oacc[t].f[r] *= scl[r];

    // transpose P through per-wave LDS into A-fragment layout
    unsigned short* Pw = Psm + wave * 1024;
#pragma unroll
    for (int t = 0; t < 4; ++t)
#pragma unroll
      for (int r = 0; r < 8; ++r)
        Pw[(r + 8 * lhi) * 64 + t * 16 + lrow] = f2bf(sacc[t].f[r]);
    asm volatile("s_wait_dscnt 0x0" ::: "memory");
    BFrag pf[2];
#pragma unroll
    for (int c = 0; c < 2; ++c) {
      const int kb = c * 32 + lhi * 8;
#pragma unroll
      for (int e = 0; e < 8; ++e) {
        pf[c].u[e]     = Pw[lrow * 64 + kb + e];
        pf[c].u[e + 8] = Pw[lrow * 64 + kb + 16 + e];
      }
    }

    // O += P @ V : V^T tile [dim][key] gives contiguous B-fragment gathers
#pragma unroll
    for (int t = 0; t < 4; ++t)
#pragma unroll
      for (int c = 0; c < 2; ++c) {
        BFrag vf;
        const int drow = t * 16 + lrow, kbb = c * 32 + lhi * 8;
#pragma unroll
        for (int e = 0; e < 8; ++e) {
          vf.u[e]     = Vt[drow * 64 + kbb + e];
          vf.u[e + 8] = Vt[drow * 64 + kbb + 16 + e];
        }
        oacc[t].v = __builtin_amdgcn_wmma_f32_16x16x32_bf16(
            false, pf[c].v, false, vf.v, (short)0, oacc[t].v, false, false);
      }
  }

  // finalize and write token-major bf16 attn output [B*NQ][H*DH]
#pragma unroll
  for (int t = 0; t < 4; ++t)
#pragma unroll
    for (int r = 0; r < 8; ++r) {
      const int q = q0 + r + 8 * lhi;
      Oattn[((size_t)b * NQ + q) * (HH * DH) + h * DH + t * 16 + lrow] =
          f2bf(oacc[t].f[r] / lsum[r]);
    }
}

// ---------------------------------------------------------------------------
extern "C" void kernel_launch(void* const* d_in, const int* in_sizes, int n_in,
                              void* d_out, int out_size, void* d_ws, size_t ws_size,
                              hipStream_t stream) {
  const float* x_query   = (const float*)d_in[0];
  const float* x_context = (const float*)d_in[1];
  const float* ropeQ = (const float*)d_in[2];
  const float* ropeC = (const float*)d_in[3];
  const float* ln_q_w = (const float*)d_in[4];
  const float* ln_q_b = (const float*)d_in[5];
  const float* ln_c_w = (const float*)d_in[6];
  const float* ln_c_b = (const float*)d_in[7];
  const float* Wq   = (const float*)d_in[8];
  const float* Wkv  = (const float*)d_in[9];
  const float* Wout = (const float*)d_in[10];
  const float* bout = (const float*)d_in[11];
  float* out = (float*)d_out;

  char* ws = (char*)d_ws;
  size_t off = 0;
  auto carve = [&](size_t bytes) { char* p = ws + off; off += (bytes + 255) & ~size_t(255); return p; };
  unsigned short* xq_bf   = (unsigned short*)carve((size_t)BB*NQ*DD*2);
  unsigned short* xc_bf   = (unsigned short*)carve((size_t)BB*NK*DD*2);
  unsigned short* wq_bf   = (unsigned short*)carve((size_t)DD*HH*DH*2);
  unsigned short* wkv_bf  = (unsigned short*)carve((size_t)DD*2*HH*DH*2);
  unsigned short* wout_bf = (unsigned short*)carve((size_t)HH*DH*DD*2);
  unsigned short* Qbuf    = (unsigned short*)carve((size_t)BB*HH*NQ*DH*2);
  unsigned short* Kbuf    = (unsigned short*)carve((size_t)BB*HH*NK*DH*2);
  unsigned short* Vbuf    = (unsigned short*)carve((size_t)BB*HH*NK*DH*2);  // [b,h,d,tok]
  unsigned short* attn_bf = (unsigned short*)carve((size_t)BB*NQ*HH*DH*2);

  ln_bf16_kernel<<<(BB*NQ)/8, 256, 0, stream>>>(x_query, ln_q_w, ln_q_b, xq_bf);
  ln_bf16_kernel<<<(BB*NK)/8, 256, 0, stream>>>(x_context, ln_c_w, ln_c_b, xc_bf);

  cvt_bf16_kernel<<<(DD*HH*DH + 255)/256, 256, 0, stream>>>(Wq, wq_bf, DD*HH*DH);
  cvt_bf16_kernel<<<(DD*2*HH*DH + 255)/256, 256, 0, stream>>>(Wkv, wkv_bf, DD*2*HH*DH);
  cvt_bf16_kernel<<<(HH*DH*DD + 255)/256, 256, 0, stream>>>(Wout, wout_bf, HH*DH*DD);

  gemm_bf16_kernel<0><<<dim3((BB*NQ)/TM, (HH*DH)/TN), 256, 0, stream>>>(
      xq_bf, wq_bf, HH*DH, NQ, ropeQ, Qbuf, nullptr, nullptr, nullptr);

  gemm_bf16_kernel<1><<<dim3((BB*NK)/TM, (2*HH*DH)/TN), 256, 0, stream>>>(
      xc_bf, wkv_bf, 2*HH*DH, NK, ropeC, Kbuf, Vbuf, nullptr, nullptr);

  flash_kernel<<<dim3(BB*HH, NQ/128), 256, 0, stream>>>(Qbuf, Kbuf, Vbuf, attn_bf);

  gemm_bf16_kernel<2><<<dim3((BB*NQ)/TM, DD/TN), 256, 0, stream>>>(
      attn_bf, wout_bf, DD, NQ, nullptr, nullptr, nullptr, bout, out);
}
